// AttentionLayer_17806934409709
// MI455X (gfx1250) — matrix operand order repaired and checked by reading, not compiled, not run on hardware
//
#include <hip/hip_runtime.h>
#include <hip/hip_bf16.h>
#include <cstddef>

// ---------------------------------------------------------------------------
// AttentionLayer for MI455X (gfx1250, wave32, WMMA).
//   x:[2,4096,512] f32 -> RMSNorm -> QKV (bf16 WMMA) -> causal flash-attn
//   (bf16 WMMA, f32 online softmax) -> out-proj (bf16 WMMA) + residual -> f32
// Uses GLOBAL_LOAD_ASYNC_TO_LDS (ASYNCcnt) + double-buffered LDS when the
// toolchain exposes the builtins; falls back to sync staging otherwise.
// ---------------------------------------------------------------------------

#define DIMC 512
#define NH   8
#define HD   64
#define NB   2
#define TT   4096
#define BT   (NB * TT)     // 8192 rows
#define N3C  (3 * DIMC)    // 1536

#if __has_builtin(__builtin_amdgcn_global_load_async_to_lds_b128) && \
    __has_builtin(__builtin_amdgcn_s_wait_asynccnt)
#define USE_ASYNC 1
#else
#define USE_ASYNC 0
#endif

typedef __attribute__((ext_vector_type(16))) __bf16 v16bf;
typedef __attribute__((ext_vector_type(8)))  float  v8f;
typedef __attribute__((ext_vector_type(8)))  __bf16 bf16x8;
typedef __attribute__((ext_vector_type(4)))  int    i32x4;

static __device__ __forceinline__ __bf16 f2bf(float f) { return (__bf16)f; }

#if USE_ASYNC
// 16B global -> LDS async copy (tracked by ASYNCcnt, no VGPR data path).
// Builtin signature on this toolchain takes v4i pointers (per diagnostic).
static __device__ __forceinline__ void async_cp16(const __bf16* g, __bf16* l) {
  __builtin_amdgcn_global_load_async_to_lds_b128(
      (__attribute__((address_space(1))) i32x4*)g,
      (__attribute__((address_space(3))) i32x4*)l, 0, 0);
}
#endif

// ---- WMMA fragment loaders (wave32, contiguous k order used by A and B alike) --------------
// A (16x32, MxK): lane m = lane&15; half i -> K = g*16 + i, g = lane>>4.
static __device__ __forceinline__ v16bf frag_A(const __bf16* base, int ld,
                                               int lane, int koff) {
  int m = lane & 15, g = lane >> 4;
  const __bf16* p = base + m * ld + koff + g * 16;
  v16bf a;
#pragma unroll
  for (int i = 0; i < 8; ++i) a[i] = p[i];
#pragma unroll
  for (int i = 0; i < 8; ++i) a[8 + i] = p[8 + i];
  return a;
}

// B (32x16, KxN) from N-major storage base[n*ld + k]: lane n = lane&15,
// half i -> K = g*16 + i.
static __device__ __forceinline__ v16bf frag_Bn(const __bf16* base, int ld,
                                                int lane, int koff) {
  int n = lane & 15, g = lane >> 4;
  const __bf16* p = base + n * ld + koff + g * 16;
  v16bf b;
#pragma unroll
  for (int i = 0; i < 16; ++i) b[i] = p[i];
  return b;
}

// B (32x16, KxN) from K-major storage base[k*ld + n] (e.g. V tile).
static __device__ __forceinline__ v16bf frag_Bk(const __bf16* base, int ld,
                                                int lane, int koff) {
  int n = lane & 15, g = lane >> 4;
  const __bf16* p = base + (koff + g * 16) * ld + n;
  v16bf b;
#pragma unroll
  for (int i = 0; i < 16; ++i) b[i] = p[i * ld];
  return b;
}

static __device__ __forceinline__ v8f wmma_bf16(v16bf a, v16bf b, v8f c) {
  return __builtin_amdgcn_wmma_f32_16x16x32_bf16(false, a, false, b, (short)0,
                                                 c, false, false);
}

// ---- Kernel: f32 -> bf16 convert -------------------------------------------
__global__ __launch_bounds__(256) void convert_bf16_kernel(
    const float* __restrict__ src, __bf16* __restrict__ dst, int n) {
  int i = (blockIdx.x * blockDim.x + threadIdx.x) * 4;
  if (i + 3 >= n) return;
  float4 v = *(const float4*)(src + i);
  dst[i + 0] = f2bf(v.x);
  dst[i + 1] = f2bf(v.y);
  dst[i + 2] = f2bf(v.z);
  dst[i + 3] = f2bf(v.w);
}

// ---- Kernel: RMSNorm, one block (128 thr) per row --------------------------
__global__ __launch_bounds__(128) void rmsnorm_kernel(
    const float* __restrict__ x, const float* __restrict__ scale,
    __bf16* __restrict__ xn) {
  const int row = blockIdx.x;
  const int tid = threadIdx.x;
  const float* xr = x + (size_t)row * DIMC;
  float4 v = *(const float4*)(xr + tid * 4);
  float ss = v.x * v.x + v.y * v.y + v.z * v.z + v.w * v.w;
#pragma unroll
  for (int off = 16; off > 0; off >>= 1) ss += __shfl_xor(ss, off, 32);
  __shared__ float red[4];
  if ((tid & 31) == 0) red[tid >> 5] = ss;
  __syncthreads();
  float tot = red[0] + red[1] + red[2] + red[3];
  float inv = rsqrtf(tot * (1.0f / DIMC) + 1e-6f);
  float4 s4 = *(const float4*)(scale + tid * 4);
  __bf16* o = xn + (size_t)row * DIMC + tid * 4;
  o[0] = f2bf(v.x * inv * s4.x);
  o[1] = f2bf(v.y * inv * s4.y);
  o[2] = f2bf(v.z * inv * s4.z);
  o[3] = f2bf(v.w * inv * s4.w);
}

// ---- Shared GEMM mainloop: 128(M) x 64(N) block tile, K in 32-steps --------
// A: [..][lda] row-major bf16 (activations), W: [N][ldw] row-major bf16
// (torch Linear weight, so B(k,n) = W[n][k]). 256 threads = 8 waves; wave w
// owns rows [w*16, w*16+16) x all 64 cols -> acc[4] f32 frags.
static __device__ __forceinline__ void gemm_mainloop_128x64(
    const __bf16* __restrict__ A, int lda, const __bf16* __restrict__ W,
    int ldw, int row0, int n0, int K, v8f acc[4]) {
  const int tid = threadIdx.x;
  const int lane = tid & 31;
  const int w = tid >> 5;
#if USE_ASYNC
  __shared__ __bf16 As[2][128 * 48];  // stride 48 bf16 = 96B, 16B-aligned rows
  __shared__ __bf16 Bs[2][64 * 48];
  const int ra0 = tid >> 2, ca = (tid & 3) * 8;   // A chunk 0
  const int ra1 = (tid + 256) >> 2;               // A chunk 1 (same ca)
  const int rb = tid >> 2;                        // W chunk
  auto issue = [&](int k0, int buf) {
    async_cp16(A + (size_t)(row0 + ra0) * lda + k0 + ca, &As[buf][ra0 * 48 + ca]);
    async_cp16(A + (size_t)(row0 + ra1) * lda + k0 + ca, &As[buf][ra1 * 48 + ca]);
    async_cp16(W + (size_t)(n0 + rb) * ldw + k0 + ca, &Bs[buf][rb * 48 + ca]);
  };
  issue(0, 0);
  __builtin_amdgcn_s_wait_asynccnt(0);
  __syncthreads();
  for (int k0 = 0; k0 < K; k0 += 32) {
    const int buf = (k0 >> 5) & 1;
    if (k0 + 32 < K) issue(k0 + 32, buf ^ 1);  // DMA next tile during compute
    v16bf a  = frag_A(&As[buf][(w * 16) * 48], 48, lane, 0);
    v16bf b0 = frag_Bn(&Bs[buf][0 * 48], 48, lane, 0);
    v16bf b1 = frag_Bn(&Bs[buf][16 * 48], 48, lane, 0);
    v16bf b2 = frag_Bn(&Bs[buf][32 * 48], 48, lane, 0);
    v16bf b3 = frag_Bn(&Bs[buf][48 * 48], 48, lane, 0);
    acc[0] = wmma_bf16(a, b0, acc[0]);
    acc[1] = wmma_bf16(a, b1, acc[1]);
    acc[2] = wmma_bf16(a, b2, acc[2]);
    acc[3] = wmma_bf16(a, b3, acc[3]);
    __builtin_amdgcn_s_wait_asynccnt(0);
    __syncthreads();
  }
#else
  __shared__ __bf16 As[128 * 48];
  __shared__ __bf16 Bs[64 * 48];
  for (int k0 = 0; k0 < K; k0 += 32) {
    __syncthreads();
#pragma unroll
    for (int i = 0; i < 2; ++i) {  // A tile: 128x32 bf16, 512 8-elem chunks
      int c = tid + i * 256;
      int r = c >> 2, co = (c & 3) * 8;
      *(bf16x8*)&As[r * 48 + co] =
          *(const bf16x8*)(A + (size_t)(row0 + r) * lda + k0 + co);
    }
    {  // W tile: 64x32 bf16, 256 chunks
      int r = tid >> 2, co = (tid & 3) * 8;
      *(bf16x8*)&Bs[r * 48 + co] =
          *(const bf16x8*)(W + (size_t)(n0 + r) * ldw + k0 + co);
    }
    __syncthreads();
    v16bf a  = frag_A(&As[(w * 16) * 48], 48, lane, 0);
    v16bf b0 = frag_Bn(&Bs[0 * 48], 48, lane, 0);
    v16bf b1 = frag_Bn(&Bs[16 * 48], 48, lane, 0);
    v16bf b2 = frag_Bn(&Bs[32 * 48], 48, lane, 0);
    v16bf b3 = frag_Bn(&Bs[48 * 48], 48, lane, 0);
    acc[0] = wmma_bf16(a, b0, acc[0]);
    acc[1] = wmma_bf16(a, b1, acc[1]);
    acc[2] = wmma_bf16(a, b2, acc[2]);
    acc[3] = wmma_bf16(a, b3, acc[3]);
  }
  __syncthreads();
#endif
}

// ---- Kernel: QKV projection, scatter to head-major Q/K/V (bf16) ------------
__global__ __launch_bounds__(256) void qkv_gemm_kernel(
    const __bf16* __restrict__ xn, const __bf16* __restrict__ wq,
    __bf16* __restrict__ qb, __bf16* __restrict__ kb,
    __bf16* __restrict__ vb) {
  const int n0 = blockIdx.x * 64;
  const int row0 = blockIdx.y * 128;
  const int lane = threadIdx.x & 31;
  const int w = threadIdx.x >> 5;
  v8f acc[4];
#pragma unroll
  for (int f = 0; f < 4; ++f)
#pragma unroll
    for (int r = 0; r < 8; ++r) acc[f][r] = 0.0f;
  gemm_mainloop_128x64(xn, DIMC, wq, DIMC, row0, n0, DIMC, acc);
  // C/D layout: lane col = lane&15; rows = r + 8*(lane>=16).
#pragma unroll
  for (int f = 0; f < 4; ++f) {
    int cg = n0 + f * 16 + (lane & 15);  // 0..1535
    int sect = cg >> 9;                  // 0=Q 1=K 2=V
    int cc = cg & 511;
    int h = cc >> 6, d = cc & 63;
    __bf16* dst = (sect == 0) ? qb : (sect == 1) ? kb : vb;
    float scl = (sect == 0) ? 0.125f : 1.0f;  // fold HEAD_DIM^-0.5 into Q
#pragma unroll
    for (int r = 0; r < 8; ++r) {
      int mrow = row0 + w * 16 + r + ((lane >> 4) << 3);
      int b = mrow >> 12, t = mrow & (TT - 1);
      dst[(((size_t)(b * NH + h)) * TT + t) * HD + d] = f2bf(acc[f][r] * scl);
    }
  }
}

// ---- Kernel: causal flash attention ----------------------------------------
// grid = (T/64 q-tiles, B*H). 128 threads = 4 waves; wave w owns q rows
// [qt*64 + w*16, +16). K/V staged 64x64 per kv-tile through (double-buffered)
// LDS; P bounced through wave-private LDS to reshape C-layout -> A-layout.
__global__ __launch_bounds__(128) void flash_attn_kernel(
    const __bf16* __restrict__ qg, const __bf16* __restrict__ kg,
    const __bf16* __restrict__ vg, __bf16* __restrict__ attn) {
  const int qt = blockIdx.x;
  const int bh = blockIdx.y;
  const int tid = threadIdx.x;
  const int lane = tid & 31;
  const int w = tid >> 5;
#if USE_ASYNC
  __shared__ __bf16 Ks[2][64 * 72];   // [kv][d], stride 72
  __shared__ __bf16 Vs[2][64 * 72];
#else
  __shared__ __bf16 Ks[1][64 * 72];
  __shared__ __bf16 Vs[1][64 * 72];
#endif
  __shared__ __bf16 Ps[4 * 16 * 72];  // per-wave P tile [16 q][64 kv]

  const __bf16* qbase = qg + (size_t)bh * TT * HD;
  const __bf16* kbase = kg + (size_t)bh * TT * HD;
  const __bf16* vbase = vg + (size_t)bh * TT * HD;

  // Q fragments live in registers for the whole kv loop (already * d^-1/2).
  const __bf16* qtile = qbase + (size_t)(qt * 64 + w * 16) * HD;
  v16bf qa0 = frag_A(qtile, HD, lane, 0);
  v16bf qa1 = frag_A(qtile, HD, lane, 32);

  v8f o[4];
  float mrow[8], lrow[8];
#pragma unroll
  for (int f = 0; f < 4; ++f)
#pragma unroll
    for (int r = 0; r < 8; ++r) o[f][r] = 0.0f;
#pragma unroll
  for (int r = 0; r < 8; ++r) { mrow[r] = -__builtin_inff(); lrow[r] = 0.0f; }

#if USE_ASYNC
  auto issueKV = [&](int kt, int buf) {
#pragma unroll
    for (int i = 0; i < 4; ++i) {  // 64x64 bf16 each, 512 16B chunks / matrix
      int c = tid + i * 128;
      int r = c >> 3, co = (c & 7) * 8;
      async_cp16(kbase + (size_t)(kt * 64 + r) * HD + co, &Ks[buf][r * 72 + co]);
      async_cp16(vbase + (size_t)(kt * 64 + r) * HD + co, &Vs[buf][r * 72 + co]);
    }
  };
  issueKV(0, 0);
  __builtin_amdgcn_s_wait_asynccnt(0);
  __syncthreads();
#endif

  for (int kt = 0; kt <= qt; ++kt) {
#if USE_ASYNC
    const int buf = kt & 1;
    if (kt < qt) issueKV(kt + 1, buf ^ 1);  // DMA next K/V during compute
#else
    const int buf = 0;
    __syncthreads();
#pragma unroll
    for (int i = 0; i < 4; ++i) {
      int c = tid + i * 128;
      int r = c >> 3, co = (c & 7) * 8;
      *(bf16x8*)&Ks[0][r * 72 + co] =
          *(const bf16x8*)(kbase + (size_t)(kt * 64 + r) * HD + co);
      *(bf16x8*)&Vs[0][r * 72 + co] =
          *(const bf16x8*)(vbase + (size_t)(kt * 64 + r) * HD + co);
    }
    __syncthreads();
#endif

    // S = Q * K^T : B(kdim=d, n=kv) = K[kv][d] -> n-major reads of Ks.
    v8f s[4];
#pragma unroll
    for (int f = 0; f < 4; ++f)
#pragma unroll
      for (int r = 0; r < 8; ++r) s[f][r] = 0.0f;
    {
      v16bf b0 = frag_Bn(&Ks[buf][0 * 72], 72, lane, 0);
      v16bf b1 = frag_Bn(&Ks[buf][16 * 72], 72, lane, 0);
      v16bf b2 = frag_Bn(&Ks[buf][32 * 72], 72, lane, 0);
      v16bf b3 = frag_Bn(&Ks[buf][48 * 72], 72, lane, 0);
      s[0] = wmma_bf16(qa0, b0, s[0]);
      s[1] = wmma_bf16(qa0, b1, s[1]);
      s[2] = wmma_bf16(qa0, b2, s[2]);
      s[3] = wmma_bf16(qa0, b3, s[3]);
      b0 = frag_Bn(&Ks[buf][0 * 72], 72, lane, 32);
      b1 = frag_Bn(&Ks[buf][16 * 72], 72, lane, 32);
      b2 = frag_Bn(&Ks[buf][32 * 72], 72, lane, 32);
      b3 = frag_Bn(&Ks[buf][48 * 72], 72, lane, 32);
      s[0] = wmma_bf16(qa1, b0, s[0]);
      s[1] = wmma_bf16(qa1, b1, s[1]);
      s[2] = wmma_bf16(qa1, b2, s[2]);
      s[3] = wmma_bf16(qa1, b3, s[3]);
    }

    if (kt == qt) {  // causal mask on the diagonal tile only
#pragma unroll
      for (int f = 0; f < 4; ++f) {
        int kvcol = f * 16 + (lane & 15);
#pragma unroll
        for (int r = 0; r < 8; ++r) {
          int qloc = w * 16 + r + ((lane >> 4) << 3);
          if (kvcol > qloc) s[f][r] = -__builtin_inff();
        }
      }
    }

    // Online softmax. Rows live in 16-lane half-waves (xor 1,2,4,8 stays
    // inside the half), so shfl reductions give per-row max/sum.
    float alpha[8];
#pragma unroll
    for (int r = 0; r < 8; ++r) {
      float rm = fmaxf(fmaxf(s[0][r], s[1][r]), fmaxf(s[2][r], s[3][r]));
#pragma unroll
      for (int off = 1; off < 16; off <<= 1) rm = fmaxf(rm, __shfl_xor(rm, off, 32));
      float mnew = fmaxf(mrow[r], rm);
      alpha[r] = __expf(mrow[r] - mnew);
      mrow[r] = mnew;
    }
#pragma unroll
    for (int f = 0; f < 4; ++f)
#pragma unroll
      for (int r = 0; r < 8; ++r) s[f][r] = __expf(s[f][r] - mrow[r]);
#pragma unroll
    for (int r = 0; r < 8; ++r) {
      float rs = s[0][r] + s[1][r] + s[2][r] + s[3][r];
#pragma unroll
      for (int off = 1; off < 16; off <<= 1) rs += __shfl_xor(rs, off, 32);
      lrow[r] = lrow[r] * alpha[r] + rs;
#pragma unroll
      for (int f = 0; f < 4; ++f) o[f][r] *= alpha[r];
    }

    // P (C-layout) -> wave-private LDS -> A-layout for P*V. Same-wave DS ops
    // are in-order (ISA 7.1), so no block barrier is needed here.
    __bf16* pw = &Ps[(w * 16) * 72];
#pragma unroll
    for (int f = 0; f < 4; ++f) {
      int col = f * 16 + (lane & 15);
#pragma unroll
      for (int r = 0; r < 8; ++r)
        pw[(r + ((lane >> 4) << 3)) * 72 + col] = f2bf(s[f][r]);
    }

#pragma unroll
    for (int ks = 0; ks < 2; ++ks) {
      v16bf pa = frag_A(pw, 72, lane, ks * 32);
      v16bf v0 = frag_Bk(&Vs[buf][0], 72, lane, ks * 32);
      v16bf v1 = frag_Bk(&Vs[buf][16], 72, lane, ks * 32);
      v16bf v2 = frag_Bk(&Vs[buf][32], 72, lane, ks * 32);
      v16bf v3 = frag_Bk(&Vs[buf][48], 72, lane, ks * 32);
      o[0] = wmma_bf16(pa, v0, o[0]);
      o[1] = wmma_bf16(pa, v1, o[1]);
      o[2] = wmma_bf16(pa, v2, o[2]);
      o[3] = wmma_bf16(pa, v3, o[3]);
    }

#if USE_ASYNC
    __builtin_amdgcn_s_wait_asynccnt(0);
    __syncthreads();
#endif
  }

  // Epilogue: O /= l, write bf16 into [BT][512] (t-major, h*64+d cols).
  const int b = bh >> 3, h = bh & 7;
#pragma unroll
  for (int r = 0; r < 8; ++r) lrow[r] = 1.0f / lrow[r];
#pragma unroll
  for (int f = 0; f < 4; ++f) {
    int d = f * 16 + (lane & 15);
#pragma unroll
    for (int r = 0; r < 8; ++r) {
      int t = qt * 64 + w * 16 + r + ((lane >> 4) << 3);
      attn[((size_t)(b * TT + t)) * DIMC + h * HD + d] = f2bf(o[f][r] * lrow[r]);
    }
  }
}

// ---- Kernel: output projection + residual (f32 out) ------------------------
__global__ __launch_bounds__(256) void proj_gemm_kernel(
    const __bf16* __restrict__ attnb, const __bf16* __restrict__ wp,
    const float* __restrict__ x, float* __restrict__ out) {
  const int n0 = blockIdx.x * 64;
  const int row0 = blockIdx.y * 128;
  const int lane = threadIdx.x & 31;
  const int w = threadIdx.x >> 5;
  v8f acc[4];
#pragma unroll
  for (int f = 0; f < 4; ++f)
#pragma unroll
    for (int r = 0; r < 8; ++r) acc[f][r] = 0.0f;
  gemm_mainloop_128x64(attnb, DIMC, wp, DIMC, row0, n0, DIMC, acc);
#pragma unroll
  for (int f = 0; f < 4; ++f) {
    int cg = n0 + f * 16 + (lane & 15);
#pragma unroll
    for (int r = 0; r < 8; ++r) {
      int mrow = row0 + w * 16 + r + ((lane >> 4) << 3);
      size_t idx = (size_t)mrow * DIMC + cg;
      out[idx] = x[idx] + acc[f][r];
    }
  }
}

// ---------------------------------------------------------------------------
extern "C" void kernel_launch(void* const* d_in, const int* in_sizes, int n_in,
                              void* d_out, int out_size, void* d_ws,
                              size_t ws_size, hipStream_t stream) {
  (void)in_sizes; (void)n_in; (void)out_size; (void)ws_size;
  const float* x = (const float*)d_in[0];
  const float* w_qkv = (const float*)d_in[1];
  const float* w_proj = (const float*)d_in[2];
  const float* norm_scale = (const float*)d_in[3];
  float* out = (float*)d_out;

  char* ws = (char*)d_ws;
  size_t off = 0;
  auto alloc = [&](size_t bytes) -> char* {
    char* p = ws + off;
    off += (bytes + 255) & ~(size_t)255;
    return p;
  };
  __bf16* xn     = (__bf16*)alloc((size_t)BT * DIMC * 2);          // 8 MB
  __bf16* wqkvb  = (__bf16*)alloc((size_t)N3C * DIMC * 2);         // 1.5 MB
  __bf16* wprojb = (__bf16*)alloc((size_t)DIMC * DIMC * 2);        // 0.5 MB
  __bf16* qb     = (__bf16*)alloc((size_t)NB * NH * TT * HD * 2);  // 8 MB
  __bf16* kb     = (__bf16*)alloc((size_t)NB * NH * TT * HD * 2);  // 8 MB
  __bf16* vb     = (__bf16*)alloc((size_t)NB * NH * TT * HD * 2);  // 8 MB
  __bf16* attnb  = (__bf16*)alloc((size_t)BT * DIMC * 2);          // 8 MB
  // total ~42 MB of d_ws

  convert_bf16_kernel<<<(N3C * DIMC) / 1024, 256, 0, stream>>>(w_qkv, wqkvb,
                                                               N3C * DIMC);
  convert_bf16_kernel<<<(DIMC * DIMC) / 1024, 256, 0, stream>>>(w_proj, wprojb,
                                                                DIMC * DIMC);
  rmsnorm_kernel<<<BT, 128, 0, stream>>>(x, norm_scale, xn);
  qkv_gemm_kernel<<<dim3(N3C / 64, BT / 128), 256, 0, stream>>>(xn, wqkvb, qb,
                                                                kb, vb);
  flash_attn_kernel<<<dim3(TT / 64, NB * NH), 128, 0, stream>>>(qb, kb, vb,
                                                                attnb);
  proj_gemm_kernel<<<dim3(DIMC / 64, BT / 128), 256, 0, stream>>>(attnb, wprojb,
                                                                  x, out);
}
